// CustomMultiHeadSelfAttention_20856361190136
// MI455X (gfx1250) — matrix operand order repaired and checked by reading, not compiled, not run on hardware
//
#include <hip/hip_runtime.h>
#include <hip/hip_bf16.h>
#include <stdint.h>

#define B_ 4
#define S_ 2048
#define E_ 1024
#define H_ 16
#define D_ 64
#define M_ (B_ * S_)   // 8192 rows

typedef __bf16 bf16;
typedef __attribute__((ext_vector_type(16))) __bf16 v16bf;
typedef __attribute__((ext_vector_type(8)))  float  v8f;
typedef int v4i __attribute__((__vector_size__(16)));   // matches builtin param

#define GLOBAL_AS __attribute__((address_space(1)))
#define LDS_AS    __attribute__((address_space(3)))

#if defined(__has_builtin)
#if __has_builtin(__builtin_amdgcn_global_load_async_to_lds_b128)
#define HAVE_ASYNC_LDS 1
#endif
#endif
#ifndef HAVE_ASYNC_LDS
#define HAVE_ASYNC_LDS 0
#endif

// 16-byte global->LDS copy. On gfx1250 with the async builtin available this
// lowers to global_load_async_to_lds_b128 (tracked on ASYNCcnt, bypasses VGPRs).
__device__ __forceinline__ void copy16_g2l(const bf16* g, bf16* l) {
#if HAVE_ASYNC_LDS
  __builtin_amdgcn_global_load_async_to_lds_b128(
      (GLOBAL_AS v4i*)const_cast<bf16*>(g), (LDS_AS v4i*)l, 0, 0);
#else
  *(uint4*)l = *(const uint4*)g;
#endif
}

__device__ __forceinline__ void async_drain() {
#if HAVE_ASYNC_LDS
#if __has_builtin(__builtin_amdgcn_s_wait_asynccnt)
  __builtin_amdgcn_s_wait_asynccnt(0);
#else
  asm volatile("s_wait_asynccnt 0" ::: "memory");
#endif
#endif
}

__device__ __forceinline__ v8f zero8() {
  v8f z = {0.f, 0.f, 0.f, 0.f, 0.f, 0.f, 0.f, 0.f};
  return z;
}

__device__ __forceinline__ v8f wmma_bf16(v16bf a, v16bf b, v8f c) {
  return __builtin_amdgcn_wmma_f32_16x16x32_bf16(
      /*neg_a=*/false, a, /*neg_b=*/false, b,
      /*c_mod=*/(short)0, c, /*reuse_a=*/false, /*reuse_b=*/false);
}

// A fragment: 16(M) x 32(K) tile, row-major with 'stride' elements.
// Lane L<16: row=L, K = 0..7 (v0-3), 16..23 (v4-7); L>=16: row=L-16, K=8..15, 24..31.
__device__ __forceinline__ v16bf load_a_frag(const bf16* tile, int stride) {
  const int lane = threadIdx.x & 31;
  const bf16* p = tile + (lane & 15) * stride + ((lane >> 4) << 3);
  v16bf r;
#pragma unroll
  for (int i = 0; i < 8; ++i) r[i] = p[i];
#pragma unroll
  for (int i = 0; i < 8; ++i) r[8 + i] = p[16 + i];
  return r;
}

// B fragment: logical K(32) x N(16), stored N-major (tile[n*stride + k]).
// Lane L<16: col=L, K=0..7, 16..23; L>=16: col=L-16, K=8..15, 24..31.
__device__ __forceinline__ v16bf load_bt_frag(const bf16* tile, int stride) {
  const int lane = threadIdx.x & 31;
  const bf16* p = tile + (lane & 15) * stride + ((lane >> 4) << 3);
  v16bf r;
#pragma unroll
  for (int i = 0; i < 8; ++i) r[i] = p[i];
#pragma unroll
  for (int i = 0; i < 8; ++i) r[8 + i] = p[16 + i];
  return r;
}

#define ASTR 40   // 32 + 8 pad; 80 bytes/row keeps 16B alignment

// ---------------- Kernel 1: in_proj + split to Q/K/V head layout -------------
__global__ __launch_bounds__(256) void inproj_kernel(
    const float* __restrict__ qkv, const float* __restrict__ in_w,
    const float* __restrict__ in_b, bf16* __restrict__ Qb,
    bf16* __restrict__ Kb, bf16* __restrict__ Vb) {
  __shared__ bf16 lA[128 * ASTR];  // X tile  [m][k]
  __shared__ bf16 lB[128 * ASTR];  // W tile  [f][k]  (N-major)

  const int tid  = threadIdx.x;
  const int wave = tid >> 5;
  const int lane = tid & 31;
  const int m0 = blockIdx.y * 128;
  const int f0 = blockIdx.x * 128;

  v8f acc[8];
#pragma unroll
  for (int j = 0; j < 8; ++j) acc[j] = zero8();

  const int lr = tid >> 3;        // 0..31
  const int lc = (tid & 7) * 4;   // 0..28 step 4 (float4 per thread)

  for (int k0 = 0; k0 < E_; k0 += 32) {
    if (k0 + 32 < E_) {  // prefetch next K-tile (global_prefetch_b8)
      __builtin_prefetch(qkv + (size_t)(m0 + lr) * E_ + k0 + 32 + lc, 0, 1);
      __builtin_prefetch(in_w + (size_t)(f0 + lr) * E_ + k0 + 32 + lc, 0, 1);
    }
#pragma unroll
    for (int rr = 0; rr < 128; rr += 32) {
      const int row = lr + rr;
      float4 av = *(const float4*)(qkv + (size_t)(m0 + row) * E_ + k0 + lc);
      bf16* da = lA + row * ASTR + lc;
      da[0] = (bf16)av.x; da[1] = (bf16)av.y; da[2] = (bf16)av.z; da[3] = (bf16)av.w;
      float4 bv = *(const float4*)(in_w + (size_t)(f0 + row) * E_ + k0 + lc);
      bf16* db = lB + row * ASTR + lc;
      db[0] = (bf16)bv.x; db[1] = (bf16)bv.y; db[2] = (bf16)bv.z; db[3] = (bf16)bv.w;
    }
    __syncthreads();
    v16bf a = load_a_frag(lA + (wave * 16) * ASTR, ASTR);
#pragma unroll
    for (int j = 0; j < 8; ++j) {
      v16bf b = load_bt_frag(lB + (j * 16) * ASTR, ASTR);
      acc[j] = wmma_bf16(a, b, acc[j]);
    }
    __syncthreads();
  }

  // epilogue: + bias, scatter bf16 into [B,H,S,D] per Q/K/V
  const int col = lane & 15;
  const int rowbase = (lane >> 4) * 8;
#pragma unroll
  for (int j = 0; j < 8; ++j) {
    const int f = f0 + j * 16 + col;
    const float bias = in_b[f];
    const int part = f >> 10;        // 0=q 1=k 2=v
    const int e = f & (E_ - 1);
    const int h = e >> 6;
    const int d = e & (D_ - 1);
    bf16* dst = (part == 0) ? Qb : (part == 1) ? Kb : Vb;
#pragma unroll
    for (int r = 0; r < 8; ++r) {
      const int m = m0 + wave * 16 + rowbase + r;
      const int bb = m >> 11;        // / S_
      const int s  = m & (S_ - 1);
      dst[(((size_t)bb * H_ + h) * S_ + s) * D_ + d] = (bf16)(acc[j][r] + bias);
    }
  }
}

// ---------------- Kernel 2: flash attention per (b,h) -----------------------
#define QSTR 72   // 64 + 8 pad (144 B rows, 16B aligned)
#define PSTR 72

__global__ __launch_bounds__(256) void attn_kernel(
    const bf16* __restrict__ Qb, const bf16* __restrict__ Kb,
    const bf16* __restrict__ Vb, bf16* __restrict__ Ob) {
  __shared__ bf16 lQ[128 * QSTR];      // Q tile  [q][d]
  __shared__ bf16 lK[64 * QSTR];       // K block [key][d]
  __shared__ bf16 lVt[64 * PSTR];      // V^T     [d][key]
  __shared__ bf16 lP[8][16 * PSTR];    // per-wave P staging [q][key]

  const int tid  = threadIdx.x;
  const int wave = tid >> 5;
  const int lane = tid & 31;
  const int qt = blockIdx.x;           // 0..15
  const int h  = blockIdx.y;
  const int b  = blockIdx.z;

  const size_t head_off = ((size_t)b * H_ + h) * S_ * D_;
  const bf16* Qh = Qb + head_off;
  const bf16* Kh = Kb + head_off;
  const bf16* Vh = Vb + head_off;

  const int lr  = tid >> 3;            // 0..31
  const int lc8 = (tid & 7) * 8;       // 8 bf16 (=16B) per thread

  // stage Q tile (128 x 64) — async global->LDS copy (no VGPR round-trip)
#pragma unroll
  for (int rr = 0; rr < 128; rr += 32) {
    const int row = lr + rr;
    copy16_g2l(Qh + (size_t)(qt * 128 + row) * D_ + lc8,
               lQ + row * QSTR + lc8);
  }

  float mrow[8], lsum[8];
  v8f oacc[4];
#pragma unroll
  for (int r = 0; r < 8; ++r) { mrow[r] = -3.0e38f; lsum[r] = 0.f; }
#pragma unroll
  for (int n = 0; n < 4; ++n) oacc[n] = zero8();

  const int col = lane & 15;
  const int rowbase = (lane >> 4) * 8;

  for (int kb = 0; kb < S_ / 64; ++kb) {
    // stage K block (async copy) and transposed V block
#pragma unroll
    for (int rr = 0; rr < 64; rr += 32) {
      const int key = lr + rr;
      copy16_g2l(Kh + (size_t)(kb * 64 + key) * D_ + lc8,
                 lK + key * QSTR + lc8);
      uint4 vv = *(const uint4*)(Vh + (size_t)(kb * 64 + key) * D_ + lc8);
      const bf16* vp = (const bf16*)&vv;
#pragma unroll
      for (int i = 0; i < 8; ++i) lVt[(lc8 + i) * PSTR + key] = vp[i];
    }
    async_drain();      // ASYNCcnt -> 0 before the barrier
    __syncthreads();

    // S = (Q K^T) * 1/sqrt(D)
    v8f sacc[4];
#pragma unroll
    for (int n = 0; n < 4; ++n) sacc[n] = zero8();
#pragma unroll
    for (int ks = 0; ks < 2; ++ks) {
      v16bf a = load_a_frag(lQ + (wave * 16) * QSTR + ks * 32, QSTR);
#pragma unroll
      for (int n = 0; n < 4; ++n) {
        v16bf bk = load_bt_frag(lK + (n * 16) * QSTR + ks * 32, QSTR);
        sacc[n] = wmma_bf16(a, bk, sacc[n]);
      }
    }
#pragma unroll
    for (int n = 0; n < 4; ++n)
#pragma unroll
      for (int r = 0; r < 8; ++r) sacc[n][r] *= 0.125f;

    // online softmax (row stats: reduce across the 16-lane column group)
#pragma unroll
    for (int r = 0; r < 8; ++r) {
      float rm = fmaxf(fmaxf(sacc[0][r], sacc[1][r]),
                       fmaxf(sacc[2][r], sacc[3][r]));
      rm = fmaxf(rm, __shfl_xor(rm, 1, 32));
      rm = fmaxf(rm, __shfl_xor(rm, 2, 32));
      rm = fmaxf(rm, __shfl_xor(rm, 4, 32));
      rm = fmaxf(rm, __shfl_xor(rm, 8, 32));
      const float mnew = fmaxf(mrow[r], rm);
      const float corr = __expf(mrow[r] - mnew);
      mrow[r] = mnew;
      float rs = 0.f;
#pragma unroll
      for (int n = 0; n < 4; ++n) {
        const float p = __expf(sacc[n][r] - mnew);
        sacc[n][r] = p;
        rs += p;
      }
      rs += __shfl_xor(rs, 1, 32);
      rs += __shfl_xor(rs, 2, 32);
      rs += __shfl_xor(rs, 4, 32);
      rs += __shfl_xor(rs, 8, 32);
      lsum[r] = lsum[r] * corr + rs;
#pragma unroll
      for (int n = 0; n < 4; ++n) oacc[n][r] *= corr;
    }

    // C-layout -> A-fragment layout via per-wave LDS staging
    bf16* Pw = lP[wave];
#pragma unroll
    for (int n = 0; n < 4; ++n)
#pragma unroll
      for (int r = 0; r < 8; ++r)
        Pw[(rowbase + r) * PSTR + n * 16 + col] = (bf16)sacc[n][r];

    // O += P @ V
#pragma unroll
    for (int ks = 0; ks < 2; ++ks) {
      v16bf a = load_a_frag(Pw + ks * 32, PSTR);
#pragma unroll
      for (int n = 0; n < 4; ++n) {
        v16bf bv = load_bt_frag(lVt + (n * 16) * PSTR + ks * 32, PSTR);
        oacc[n] = wmma_bf16(a, bv, oacc[n]);
      }
    }
    __syncthreads();
  }

  // normalize and store to [B,S,E] bf16
#pragma unroll
  for (int n = 0; n < 4; ++n) {
#pragma unroll
    for (int r = 0; r < 8; ++r) {
      const int q = qt * 128 + wave * 16 + rowbase + r;
      const int d = n * 16 + col;
      const float v = oacc[n][r] / lsum[r];
      Ob[((size_t)(b * S_ + q)) * E_ + h * D_ + d] = (bf16)v;
    }
  }
}

// ---------------- Kernel 3: out_proj ----------------------------------------
__global__ __launch_bounds__(256) void outproj_kernel(
    const bf16* __restrict__ X, const float* __restrict__ out_w,
    const float* __restrict__ out_b, float* __restrict__ out) {
  __shared__ bf16 lA[128 * ASTR];
  __shared__ bf16 lB[128 * ASTR];

  const int tid  = threadIdx.x;
  const int wave = tid >> 5;
  const int lane = tid & 31;
  const int m0 = blockIdx.y * 128;
  const int f0 = blockIdx.x * 128;

  v8f acc[8];
#pragma unroll
  for (int j = 0; j < 8; ++j) acc[j] = zero8();

  const int lr = tid >> 3;       // f32 W loads
  const int lc = (tid & 7) * 4;
  const int ar = tid >> 2;       // bf16 A loads: 0..63
  const int ac = (tid & 3) * 8;  // 0,8,16,24

  for (int k0 = 0; k0 < E_; k0 += 32) {
    if (k0 + 32 < E_) {
      __builtin_prefetch(X + (size_t)(m0 + ar) * E_ + k0 + 32 + ac, 0, 1);
      __builtin_prefetch(out_w + (size_t)(f0 + lr) * E_ + k0 + 32 + lc, 0, 1);
    }
#pragma unroll
    for (int rr = 0; rr < 128; rr += 64) {
      const int row = ar + rr;
      copy16_g2l(X + (size_t)(m0 + row) * E_ + k0 + ac, lA + row * ASTR + ac);
    }
#pragma unroll
    for (int rr = 0; rr < 128; rr += 32) {
      const int row = lr + rr;
      float4 bv = *(const float4*)(out_w + (size_t)(f0 + row) * E_ + k0 + lc);
      bf16* db = lB + row * ASTR + lc;
      db[0] = (bf16)bv.x; db[1] = (bf16)bv.y; db[2] = (bf16)bv.z; db[3] = (bf16)bv.w;
    }
    async_drain();
    __syncthreads();
    v16bf a = load_a_frag(lA + (wave * 16) * ASTR, ASTR);
#pragma unroll
    for (int j = 0; j < 8; ++j) {
      v16bf b = load_bt_frag(lB + (j * 16) * ASTR, ASTR);
      acc[j] = wmma_bf16(a, b, acc[j]);
    }
    __syncthreads();
  }

  const int col = lane & 15;
  const int rowbase = (lane >> 4) * 8;
#pragma unroll
  for (int j = 0; j < 8; ++j) {
    const int f = f0 + j * 16 + col;
    const float bias = out_b[f];
#pragma unroll
    for (int r = 0; r < 8; ++r) {
      const int m = m0 + wave * 16 + rowbase + r;
      out[(size_t)m * E_ + f] = acc[j][r] + bias;
    }
  }
}

// ---------------- launch -----------------------------------------------------
extern "C" void kernel_launch(void* const* d_in, const int* in_sizes, int n_in,
                              void* d_out, int out_size, void* d_ws, size_t ws_size,
                              hipStream_t stream) {
  const float* qkv   = (const float*)d_in[0];
  const float* in_w  = (const float*)d_in[1];
  const float* in_b  = (const float*)d_in[2];
  const float* out_w = (const float*)d_in[3];
  const float* out_b = (const float*)d_in[4];
  float* out = (float*)d_out;

  const size_t tens = (size_t)M_ * E_;   // 8M elements
  bf16* Qb = (bf16*)d_ws;
  bf16* Kb = Qb + tens;
  bf16* Vb = Kb + tens;
  bf16* Ob = Vb + tens;                  // total 64 MB of workspace

  inproj_kernel<<<dim3(3 * E_ / 128, M_ / 128), 256, 0, stream>>>(
      qkv, in_w, in_b, Qb, Kb, Vb);
  attn_kernel<<<dim3(S_ / 128, H_, B_), 256, 0, stream>>>(Qb, Kb, Vb, Ob);
  outproj_kernel<<<dim3(E_ / 128, M_ / 128), 256, 0, stream>>>(
      Ob, out_w, out_b, out);
}
